// RFNN_Decoder_64991445123989
// MI455X (gfx1250) — matrix operand order repaired and checked by reading, not compiled, not run on hardware
//
#include <hip/hip_runtime.h>

// Problem sizes (from reference)
#define SEQ      512
#define FEAT     256
#define RULES    1024
#define OUT      256

// Phase-1 geometry
#define G        32            // workgroups cooperating on the recurrence
#define TPB      256           // threads per workgroup (8 waves of 32)
#define RPW      (RULES / G)   // 32 rules owned per workgroup
#define FPT      (FEAT / 8)    // 32 features per thread (8 threads per rule)

// exp(-0.5 z^2) = exp2( z^2 * (-0.5*log2 e) )
#define KNEG     (-0.72134752044448170f)

// Workspace layout (bytes)
#define RULES16_BYTES ((size_t)SEQ * RULES * 2)          // 1 MiB   f16 normalized rules
#define FCW16_BYTES   ((size_t)OUT * RULES * 2)          // 512 KiB f16 fc_w
#define PART_BYTES    ((size_t)SEQ * G * 4)              // 64 KiB  per-step per-WG partials
#define CNT_BYTES     ((size_t)SEQ * 4)                  // 2 KiB   per-step arrival counters

typedef __attribute__((ext_vector_type(16))) _Float16     v16h;
typedef __attribute__((ext_vector_type(8)))  _Float16     v8h;
typedef __attribute__((ext_vector_type(4)))  _Float16     v4h;
typedef __attribute__((ext_vector_type(8)))  float        v8f;
typedef __attribute__((ext_vector_type(4)))  unsigned int u32x4;
typedef __attribute__((ext_vector_type(8)))  int          i32x8;
typedef __attribute__((ext_vector_type(4)))  int          i32x4;

// ---------------------------------------------------------------------------
// Init: zero per-step arrival counters (ws is NOT re-poisoned between replays).
// ---------------------------------------------------------------------------
__global__ void rfnn_zero_cnt(unsigned* __restrict__ cnt) {
  int i = blockIdx.x * blockDim.x + threadIdx.x;
  if (i < SEQ) cnt[i] = 0u;
}

// ---------------------------------------------------------------------------
// One-time f32 -> f16 conversion of fc_w (removes per-tile re-conversion and
// halves B-side bytes in the WMMA loop). 4 elements / thread, vectorized.
// ---------------------------------------------------------------------------
__global__ __launch_bounds__(256) void rfnn_cvt_fcw(
    const float* __restrict__ fc_w, _Float16* __restrict__ fcw16) {
  int i = (blockIdx.x * 256 + threadIdx.x) * 4;   // OUT*RULES / 4 = 65536 thr
  float4 f = *(const float4*)(fc_w + i);
  v4h h; h[0] = (_Float16)f.x; h[1] = (_Float16)f.y;
         h[2] = (_Float16)f.z; h[3] = (_Float16)f.w;
  *(v4h*)(fcw16 + i) = h;
}

// ---------------------------------------------------------------------------
// Phase 1: serial recurrence across SEQ steps (unchanged from round 1; it
// compiled to the intended mix). Each WG owns 32 rules, carry + parameters
// register-resident; exp2 work for step t overlaps the wait on step t-1's
// global sum; deterministic fixed-order cross-WG summation.
// ---------------------------------------------------------------------------
__global__ __launch_bounds__(TPB) void rfnn_recurrence(
    const float* __restrict__ x,
    const float* __restrict__ memory,
    const float* __restrict__ center,
    const float* __restrict__ sigma,
    _Float16*    __restrict__ rules16,
    float*       __restrict__ partials,
    unsigned*    __restrict__ cnt)
{
  const int tid = threadIdx.x;
  const int wg  = blockIdx.x;
  const int fg  = tid & 7;
  const int rl  = tid >> 3;
  const int rg  = wg * RPW + rl;

  __shared__ float xs[FEAT];
  __shared__ float wsum[TPB / 32];
  __shared__ float sShared;

  float creg[FPT], isreg[FPT];
#pragma unroll
  for (int i = 0; i < FPT; ++i) {
    const int f = fg + 8 * i;
    creg[i]  = center[(size_t)rg * FEAT + f];
    isreg[i] = 1.0f / sigma[(size_t)rg * FEAT + f];
  }
  float mem = memory[rg];
  float u   = 0.0f;

  for (int t = 0; t < SEQ; ++t) {
    __syncthreads();
    xs[tid] = x[(size_t)t * FEAT + tid];
    __syncthreads();

    float mreg[FPT];
#pragma unroll
    for (int i = 0; i < FPT; ++i) {
      const float z = (xs[fg + 8 * i] - creg[i]) * isreg[i];
      mreg[i] = exp2f(z * z * KNEG);
    }

    if (t > 0) {
      __builtin_amdgcn_s_cluster_barrier();  // NOP unless cluster-dispatched
      if (tid == 0) {
        while (__hip_atomic_load(&cnt[t - 1], __ATOMIC_ACQUIRE,
                                 __HIP_MEMORY_SCOPE_AGENT) < (unsigned)G)
          __builtin_amdgcn_s_sleep(1);
        float s = 0.0f;
        for (int g = 0; g < G; ++g) s += partials[(size_t)(t - 1) * G + g];
        sShared = s;
      }
      __syncthreads();
      const float s   = sShared;
      const float inv = (s == 0.0f) ? 1.0f : 1.0f / s;
      mem = u * inv;
      if (fg == 0) rules16[(size_t)(t - 1) * RULES + rg] = (_Float16)mem;
    }

    float p = 1.0f;
#pragma unroll
    for (int i = 0; i < FPT; ++i) p *= (mreg[i] + mem);
    p *= __shfl_xor(p, 1);
    p *= __shfl_xor(p, 2);
    p *= __shfl_xor(p, 4);
    u = p;

    float v = (fg == 0) ? u : 0.0f;
    v += __shfl_xor(v, 8);
    v += __shfl_xor(v, 16);
    if ((tid & 31) == 0) wsum[tid >> 5] = v;
    __syncthreads();
    if (tid == 0) {
      float pw = 0.0f;
#pragma unroll
      for (int w = 0; w < TPB / 32; ++w) pw += wsum[w];
      __hip_atomic_store(&partials[(size_t)t * G + wg], pw,
                         __ATOMIC_RELEASE, __HIP_MEMORY_SCOPE_AGENT);
      __hip_atomic_fetch_add(&cnt[t], 1u, __ATOMIC_RELEASE,
                             __HIP_MEMORY_SCOPE_AGENT);
    }
  }

  if (tid == 0) {
    while (__hip_atomic_load(&cnt[SEQ - 1], __ATOMIC_ACQUIRE,
                             __HIP_MEMORY_SCOPE_AGENT) < (unsigned)G)
      __builtin_amdgcn_s_sleep(1);
    float s = 0.0f;
    for (int g = 0; g < G; ++g) s += partials[(size_t)(SEQ - 1) * G + g];
    sShared = s;
  }
  __syncthreads();
  {
    const float s   = sShared;
    const float inv = (s == 0.0f) ? 1.0f : 1.0f / s;
    if (fg == 0) rules16[(size_t)(SEQ - 1) * RULES + rg] = (_Float16)(u * inv);
  }
}

// ---------------------------------------------------------------------------
// Phase 2: out[512,256] = rules16 @ fcw16^T + fc_b via WMMA f16.
// 64 blocks x 8 waves; the 8 waves of a block share ONE A tile
// (16 seq-rows x K=1024 = 32 KiB, contiguous in rules16) -> staged into LDS
// by a single Tensor Data Mover descriptor (D#, ISA ch.8), waited with
// s_wait_tensorcnt. B (f16, pre-converted) is read directly (L2-resident).
// ---------------------------------------------------------------------------
__global__ __launch_bounds__(256) void rfnn_gemm_wmma(
    const _Float16* __restrict__ rules16,  // [SEQ, RULES]
    const _Float16* __restrict__ fcw16,    // [OUT, RULES]
    const float*    __restrict__ fc_b,     // [OUT]
    float*          __restrict__ out)      // [SEQ, OUT]
{
  const int tid = threadIdx.x;
  const int l   = tid & 31;
  const int wav = tid >> 5;
  const int blk = blockIdx.x;          // 0..63
  const int tm  = blk >> 1;            // seq-tile 0..31 (shared by block)
  const int tn  = (blk & 1) * 8 + wav; // out-tile 0..15 (one per wave)

  __shared__ _Float16 As[16 * RULES];  // 32 KiB A tile

  // ---- TDM stage: one 1-D tile, 16384 x 2B elements, issued by wave 0 ----
  if (wav == 0) {
    const unsigned long long ga =
        (unsigned long long)(rules16 + (size_t)tm * 16 * RULES);
    const unsigned lds = (unsigned)(size_t)&As[0];

    u32x4 g0;
    g0[0] = 1u;                                   // count=1, is_restore=0
    g0[1] = lds;                                  // lds_addr
    g0[2] = (unsigned)(ga & 0xFFFFFFFFu);         // global_addr[31:0]
    g0[3] = (unsigned)((ga >> 32) & 0x01FFFFFFu)  // global_addr[56:32]
            | 0x80000000u;                        // type=2 ("image")

    i32x8 g1;
    g1[0] = (1 << 16);            // workgroup_mask=0, data_size=1 (2B)
    g1[1] = (16384 & 0xFFFF) << 16;               // tensor_dim0 lo16 @ [63:48]
    g1[2] = (16384 >> 16) | (1 << 16);            // dim0 hi16; tensor_dim1=1
    g1[3] = (16384 & 0xFFFF) << 16;               // tile_dim0=16384 @ [127:112]
    g1[4] = 0;                                    // tile_dim1=0, tile_dim2=0
    g1[5] = 16384;                                // tensor_dim0_stride lo32
    g1[6] = 0;
    g1[7] = 0;

    i32x4 g2 = {0, 0, 0, 0};
    i32x4 g3 = {0, 0, 0, 0};

#if __clang_major__ >= 23
    i32x8 g4 = {0, 0, 0, 0, 0, 0, 0, 0};
    __builtin_amdgcn_tensor_load_to_lds(g0, g1, g2, g3, g4, 0);
#else
    __builtin_amdgcn_tensor_load_to_lds(g0, g1, g2, g3, 0);
#endif
    __builtin_amdgcn_s_wait_tensorcnt(0);
  }
  __syncthreads();

  // ---- WMMA main loop (ISA 7.12.2 wave32 operand layouts) ----
  const int rowL = l & 15;                 // A row within tile / C column
  const int colB = tn * 16 + (l & 15);     // B column (output feature)
  const int kbA  = (l >> 4) * 8;           // A: halves 0..7 -> K=kb..kb+7,
                                           //    halves 8..15 -> K=kb+16..kb+23
  const int kbB  = (l >> 4) * 8;           // B: same K order as A

  const _Float16* __restrict__ aL  = As + (size_t)rowL * RULES;
  const _Float16* __restrict__ bG  = fcw16 + (size_t)colB * RULES;

  v8f c = {};
  for (int kt = 0; kt < RULES; kt += 32) {
    if (kt + 32 < RULES)
      __builtin_prefetch(bG + kt + 32 + kbB, 0, 1);

    const v8h a0 = *(const v8h*)(aL + kt + kbA);        // ds_load_b128
    const v8h a1 = *(const v8h*)(aL + kt + kbA + 16);
    v16h a;
#pragma unroll
    for (int j = 0; j < 8; ++j) { a[j] = a0[j]; a[8 + j] = a1[j]; }

    const v8h b0 = *(const v8h*)(bG + kt + kbB);        // global b128
    const v8h b1 = *(const v8h*)(bG + kt + kbB + 16);
    v16h b;
#pragma unroll
    for (int j = 0; j < 8; ++j) { b[j] = b0[j]; b[8 + j] = b1[j]; }

    c = __builtin_amdgcn_wmma_f32_16x16x32_f16(false, a, false, b,
                                               (short)0, c, false, false);
  }

  const float bias = fc_b[colB];
#pragma unroll
  for (int i = 0; i < 8; ++i) {
    const int r = tm * 16 + i + 8 * (l >> 4);   // C/D: VGPR i -> row i+8*(l>>4)
    out[(size_t)r * OUT + colB] = c[i] + bias;
  }
}

// ---------------------------------------------------------------------------
extern "C" void kernel_launch(void* const* d_in, const int* in_sizes, int n_in,
                              void* d_out, int out_size, void* d_ws, size_t ws_size,
                              hipStream_t stream) {
  const float* x      = (const float*)d_in[0];
  const float* memory = (const float*)d_in[1];
  const float* center = (const float*)d_in[2];
  const float* sigma  = (const float*)d_in[3];
  const float* fc_w   = (const float*)d_in[4];
  const float* fc_b   = (const float*)d_in[5];

  char* ws = (char*)d_ws;
  _Float16* rules16  = (_Float16*)(ws);
  _Float16* fcw16    = (_Float16*)(ws + RULES16_BYTES);
  float*    partials = (float*)(ws + RULES16_BYTES + FCW16_BYTES);
  unsigned* cnt      = (unsigned*)(ws + RULES16_BYTES + FCW16_BYTES + PART_BYTES);

  rfnn_zero_cnt<<<(SEQ + 255) / 256, 256, 0, stream>>>(cnt);
  rfnn_cvt_fcw<<<(OUT * RULES) / (256 * 4), 256, 0, stream>>>(fc_w, fcw16);
  rfnn_recurrence<<<G, TPB, 0, stream>>>(x, memory, center, sigma,
                                         rules16, partials, cnt);
  rfnn_gemm_wmma<<<(SEQ / 16) * (OUT / 16) / 8, 256, 0, stream>>>(
      rules16, fcw16, fc_b, (float*)d_out);
}
